// CrossAttention_76948634075402
// MI455X (gfx1250) — compile-verified, matrix operand order checked by reading, not run on hardware
//
#include <hip/hip_runtime.h>
#include <math.h>

// ---------------------------------------------------------------------------
// Cross-attention forward for MI455X (gfx1250, wave32, WMMA bf16 16x16x32).
//   Pass 1: fused Q/K/V projections. Raw f32 W slab async-double-buffered in
//           LDS (padded rows), per-wave 16x64 tile, f32->bf16 head-major ws.
//   Pass 2: flash attention. 8 waves/block share one (b,h); K/V key-blocks
//           double-buffered in padded LDS via global_load_async_to_lds_b128.
//   Pass 3: output projection, same pipeline as pass 1, f32 result.
// All LDS row strides padded to kill bank conflicts (64 banks x 4B).
// ---------------------------------------------------------------------------

typedef __attribute__((ext_vector_type(16))) __bf16 v16bf;
typedef __attribute__((ext_vector_type(8)))  float  v8f;

#define DEV __device__ __forceinline__

constexpr int Bc   = 4;
constexpr int Sc   = 2048;
constexpr int Dm   = 1024;
constexpr int Hc   = 16;
constexpr int Dh   = 64;
constexpr int Mtot = Bc * Sc;          // 8192 rows
constexpr int WPB  = 8;                // waves per block (256 threads)

constexpr int KVP = 72;                // K/V LDS row stride (bf16), padded
constexpr int PLP = 36;                // P-tile LDS row stride (bf16), padded
constexpr int WLP = 68;                // W-slab LDS row stride (f32), padded

// ---------------------------------------------------------------------------
// WMMA fragment helpers (ISA 7.12.2 16-bit operand layout).
// ---------------------------------------------------------------------------
DEV int frag_k(int g, int half) {
    return ((g & 4) << 2) + half * 8 + ((g & 3) << 1);
}

// Element (idx = lane%16, k) read from base + idx*stride_idx + k*stride_k.
DEV v16bf load_frag_f32(const float* base, int stride_idx, int stride_k) {
    const int l    = threadIdx.x & 31;
    const int half = l >> 4;
    const int idx  = l & 15;
    const float* p = base + (size_t)idx * stride_idx;
    v16bf f;
#pragma unroll
    for (int g = 0; g < 8; ++g) {
        const int k = frag_k(g, half);
        f[2 * g]     = (__bf16)p[(size_t)k * stride_k];
        f[2 * g + 1] = (__bf16)p[(size_t)(k + 1) * stride_k];
    }
    return f;
}

DEV v16bf load_frag_bf16(const __bf16* base, int stride_idx, int stride_k) {
    const int l    = threadIdx.x & 31;
    const int half = l >> 4;
    const int idx  = l & 15;
    const __bf16* p = base + (size_t)idx * stride_idx;
    v16bf f;
#pragma unroll
    for (int g = 0; g < 8; ++g) {
        const int k = frag_k(g, half);
        f[2 * g]     = p[(size_t)k * stride_k];
        f[2 * g + 1] = p[(size_t)(k + 1) * stride_k];
    }
    return f;
}

DEV v8f wmma_bf16(v16bf a, v16bf b, v8f c) {
    return __builtin_amdgcn_wmma_f32_16x16x32_bf16(
        false, a, false, b, (short)0, c, false, false);
}

// ---------------------------------------------------------------------------
// Async global -> LDS staging (CDNA5, ASYNCcnt).  Per-lane 16B chunk copies.
// ---------------------------------------------------------------------------
typedef __attribute__((address_space(3))) const void lds_cvoid;

DEV unsigned lds_addr_of(const void* p) {
    return (unsigned)(unsigned long long)(lds_cvoid*)p;
}

DEV void async_chunk16(unsigned lds_byte_off, const void* g) {
    asm volatile("global_load_async_to_lds_b128 %0, %1, off"
                 :: "v"(lds_byte_off), "v"(g)
                 : "memory");
}

DEV void wait_async0() {
    asm volatile("s_wait_asynccnt 0" ::: "memory");
}

// 32 x 64 bf16 block (contiguous in global) -> padded LDS [32][KVP].
// 256 chunks of 16B: exactly one per thread.  No divergence.
DEV void stage_kv_async(__bf16* dst, const __bf16* src) {
    const int c   = threadIdx.x;
    const int row = c >> 3, col = c & 7;
    async_chunk16(lds_addr_of(dst) + (unsigned)(row * (KVP * 2) + col * 16),
                  src + row * 64 + col * 8);
}

// 32 x 64 f32 slab of W (row stride Dm in global) -> padded LDS [32][WLP].
// 512 chunks of 16B: two per thread.  No divergence.
DEV void stage_w_async(float* dst, const float* src) {
#pragma unroll
    for (int it = 0; it < 2; ++it) {
        const int c   = threadIdx.x + it * 256;
        const int row = c >> 4, col = c & 15;
        async_chunk16(lds_addr_of(dst) + (unsigned)(row * (WLP * 4) + col * 16),
                      src + (size_t)row * Dm + col * 4);
    }
}

// ---------------------------------------------------------------------------
// Pass 1: fused QKV projection.  grid.z selects Q / K / V.
// Block: 8 waves, tile M=128 x N=64.  f32 W slab async double-buffered.
// ---------------------------------------------------------------------------
__global__ void __launch_bounds__(256)
proj_qkv_kernel(const float* __restrict__ x,  const float* __restrict__ y,
                const float* __restrict__ wq, const float* __restrict__ bq,
                const float* __restrict__ wk, const float* __restrict__ bk,
                const float* __restrict__ wv, const float* __restrict__ bv,
                __bf16* __restrict__ qws, __bf16* __restrict__ kws,
                __bf16* __restrict__ vws) {
    const float* A; const float* W; const float* bias; __bf16* out;
    if (blockIdx.z == 0)      { A = x; W = wq; bias = bq; out = qws; }
    else if (blockIdx.z == 1) { A = y; W = wk; bias = bk; out = kws; }
    else                      { A = y; W = wv; bias = bv; out = vws; }

    __shared__ float wbuf[2][32 * WLP];    // ~17KB, double-buffered W slab

    const int tid = threadIdx.x, wid = tid >> 5, l = tid & 31;
    const int mb = blockIdx.x >> 4;            // 64 m-blocks
    const int ns = blockIdx.x & 15;            // 16 n-strips
    const int m0 = mb * 128 + wid * 16;
    const int n0 = ns * 64;

    v8f c0 = {}, c1 = {}, c2 = {}, c3 = {};

    // prologue: slab for k0 = 0
    stage_w_async(wbuf[0], W + n0);
    wait_async0();
    __syncthreads();

    for (int k0 = 0; k0 < Dm; k0 += 32) {
        const int cur = (k0 >> 5) & 1, nxt = cur ^ 1;
        if (k0 + 32 < Dm) {
            stage_w_async(wbuf[nxt], W + (size_t)(k0 + 32) * Dm + n0);
            __builtin_prefetch(A + (size_t)m0 * Dm + k0 + 32, 0, 3);
        }
        const float* ws = wbuf[cur];

        const v16bf a = load_frag_f32(A + (size_t)m0 * Dm + k0, Dm, 1);
        c0 = wmma_bf16(a, load_frag_f32(ws +  0, 1, WLP), c0);
        c1 = wmma_bf16(a, load_frag_f32(ws + 16, 1, WLP), c1);
        c2 = wmma_bf16(a, load_frag_f32(ws + 32, 1, WLP), c2);
        c3 = wmma_bf16(a, load_frag_f32(ws + 48, 1, WLP), c3);

        wait_async0();      // next slab fills (issued above) complete
        __syncthreads();    // visible block-wide; cur buffer now free
    }

    // ---- store head-major [B*H][S][Dh] bf16 -------------------------------
    const int half = l >> 4, n = l & 15;
    const int b_i = m0 >> 11;
    v8f cc[4] = { c0, c1, c2, c3 };
#pragma unroll
    for (int t = 0; t < 4; ++t) {
        const int col = n0 + t * 16 + n;
        const float bval = bias[col];
        const int h  = col >> 6;
        const int dh = col & 63;
#pragma unroll
        for (int i = 0; i < 8; ++i) {
            const int m = m0 + i + 8 * half;
            const int s = m & (Sc - 1);
            out[(((size_t)(b_i * Hc + h) * Sc + s) << 6) + dh] =
                (__bf16)(cc[t][i] + bval);
        }
    }
}

// ---------------------------------------------------------------------------
// Pass 2: flash attention.  Block = 8 waves sharing one (b,h); each wave owns
// a 16-query tile.  32x64 K and V blocks double-buffered in padded LDS.
// ---------------------------------------------------------------------------
__global__ void __launch_bounds__(256)
attn_kernel(const __bf16* __restrict__ Qm, const __bf16* __restrict__ Km,
            const __bf16* __restrict__ Vm, __bf16* __restrict__ out) {
    __shared__ __bf16 kbuf[2][32 * KVP];     // padded rows: conflict-free
    __shared__ __bf16 vbuf[2][32 * KVP];
    __shared__ __bf16 plds[WPB][16 * PLP];   // per-wave P tile, padded

    const int tid  = threadIdx.x, wid = tid >> 5, l = tid & 31;
    const int half = l >> 4, n = l & 15;

    const int bh = blockIdx.x >> 4;                      // 64 (b,h) pairs
    const int q0 = ((blockIdx.x & 15) * WPB + wid) << 4; // 16-query tile

    const __bf16* q  = Qm + (size_t)bh * Sc * Dh;
    const __bf16* kp = Km + (size_t)bh * Sc * Dh;
    const __bf16* vp = Vm + (size_t)bh * Sc * Dh;

    const v16bf qa0 = load_frag_bf16(q + (size_t)q0 * Dh,      Dh, 1);
    const v16bf qa1 = load_frag_bf16(q + (size_t)q0 * Dh + 32, Dh, 1);

    // preload key-block 0
    stage_kv_async(kbuf[0], kp);
    stage_kv_async(vbuf[0], vp);
    wait_async0();
    __syncthreads();

    float mrun[8], lrun[8];
    v8f acc0 = {}, acc1 = {}, acc2 = {}, acc3 = {};
#pragma unroll
    for (int i = 0; i < 8; ++i) { mrun[i] = -3.0e38f; lrun[i] = 0.0f; }

    for (int kb = 0; kb < Sc / 32; ++kb) {
        const int cur = kb & 1, nxt = cur ^ 1;
        if (kb + 1 < Sc / 32) {          // prefetch next block into other buf
            stage_kv_async(kbuf[nxt], kp + (size_t)(kb + 1) * 32 * Dh);
            stage_kv_async(vbuf[nxt], vp + (size_t)(kb + 1) * 32 * Dh);
        }
        const __bf16* kl = kbuf[cur];
        const __bf16* vl = vbuf[cur];

        // ---- scores: 16q x 32keys, contraction over Dh=64 -----------------
        v8f s0 = {}, s1 = {};
        s0 = wmma_bf16(qa0, load_frag_bf16(kl,                KVP, 1), s0);
        s0 = wmma_bf16(qa1, load_frag_bf16(kl + 32,           KVP, 1), s0);
        s1 = wmma_bf16(qa0, load_frag_bf16(kl + 16 * KVP,      KVP, 1), s1);
        s1 = wmma_bf16(qa1, load_frag_bf16(kl + 16 * KVP + 32, KVP, 1), s1);

        // ---- online softmax -----------------------------------------------
        float bm[8], p0[8], p1[8], rs[8];
#pragma unroll
        for (int i = 0; i < 8; ++i) {
            s0[i] *= 0.125f;              // 1/sqrt(64)
            s1[i] *= 0.125f;
            bm[i] = fmaxf(s0[i], s1[i]);
        }
#pragma unroll
        for (int off = 1; off < 16; off <<= 1)
#pragma unroll
            for (int i = 0; i < 8; ++i)
                bm[i] = fmaxf(bm[i], __shfl_xor(bm[i], off, 32));
#pragma unroll
        for (int i = 0; i < 8; ++i) {
            const float mn    = fmaxf(mrun[i], bm[i]);
            const float alpha = __expf(mrun[i] - mn);
            p0[i] = __expf(s0[i] - mn);
            p1[i] = __expf(s1[i] - mn);
            rs[i] = p0[i] + p1[i];
            mrun[i] = mn;
            lrun[i] *= alpha;
            acc0[i] *= alpha; acc1[i] *= alpha;
            acc2[i] *= alpha; acc3[i] *= alpha;
        }
#pragma unroll
        for (int off = 1; off < 16; off <<= 1)
#pragma unroll
            for (int i = 0; i < 8; ++i)
                rs[i] += __shfl_xor(rs[i], off, 32);
#pragma unroll
        for (int i = 0; i < 8; ++i) lrun[i] += rs[i];

        // ---- C-layout -> A-layout through per-wave LDS tile ---------------
        __bf16* my = plds[wid];
#pragma unroll
        for (int i = 0; i < 8; ++i) {
            const int row = i + 8 * half;
            my[row * PLP + n]      = (__bf16)p0[i];
            my[row * PLP + 16 + n] = (__bf16)p1[i];
        }
        const v16bf pa = load_frag_bf16(my, PLP, 1);

        // ---- acc += P (16x32) x V (32x64) ---------------------------------
        acc0 = wmma_bf16(pa, load_frag_bf16(vl +  0, 1, KVP), acc0);
        acc1 = wmma_bf16(pa, load_frag_bf16(vl + 16, 1, KVP), acc1);
        acc2 = wmma_bf16(pa, load_frag_bf16(vl + 32, 1, KVP), acc2);
        acc3 = wmma_bf16(pa, load_frag_bf16(vl + 48, 1, KVP), acc3);

        wait_async0();        // next-buffer fills issued this iteration done
        __syncthreads();      // all waves' fills visible; cur buf free
    }

    // ---- normalize and store to [B, S, D] bf16 ----------------------------
    const int bb = bh >> 4, h = bh & 15;
#pragma unroll
    for (int i = 0; i < 8; ++i) {
        const float inv = 1.0f / lrun[i];
        const int row   = q0 + i + 8 * half;
        const size_t base = ((size_t)(bb * Sc + row) * Dm) + h * Dh + n;
        out[base +  0] = (__bf16)(acc0[i] * inv);
        out[base + 16] = (__bf16)(acc1[i] * inv);
        out[base + 32] = (__bf16)(acc2[i] * inv);
        out[base + 48] = (__bf16)(acc3[i] * inv);
    }
}

// ---------------------------------------------------------------------------
// Pass 3: output projection.  A(bf16)[8192,1024] @ wo + bo -> f32 d_out.
// Same async double-buffered pipeline as pass 1.
// ---------------------------------------------------------------------------
__global__ void __launch_bounds__(256)
oproj_kernel(const __bf16* __restrict__ A, const float* __restrict__ W,
             const float* __restrict__ bias, float* __restrict__ out) {
    __shared__ float wbuf[2][32 * WLP];

    const int tid = threadIdx.x, wid = tid >> 5, l = tid & 31;
    const int mb = blockIdx.x >> 4;
    const int ns = blockIdx.x & 15;
    const int m0 = mb * 128 + wid * 16;
    const int n0 = ns * 64;

    v8f c0 = {}, c1 = {}, c2 = {}, c3 = {};

    stage_w_async(wbuf[0], W + n0);
    wait_async0();
    __syncthreads();

    for (int k0 = 0; k0 < Dm; k0 += 32) {
        const int cur = (k0 >> 5) & 1, nxt = cur ^ 1;
        if (k0 + 32 < Dm) {
            stage_w_async(wbuf[nxt], W + (size_t)(k0 + 32) * Dm + n0);
            __builtin_prefetch(A + (size_t)m0 * Dm + k0 + 32, 0, 3);
        }
        const float* ws = wbuf[cur];

        const v16bf a = load_frag_bf16(A + (size_t)m0 * Dm + k0, Dm, 1);
        c0 = wmma_bf16(a, load_frag_f32(ws +  0, 1, WLP), c0);
        c1 = wmma_bf16(a, load_frag_f32(ws + 16, 1, WLP), c1);
        c2 = wmma_bf16(a, load_frag_f32(ws + 32, 1, WLP), c2);
        c3 = wmma_bf16(a, load_frag_f32(ws + 48, 1, WLP), c3);

        wait_async0();
        __syncthreads();
    }

    const int half = l >> 4, n = l & 15;
    v8f cc[4] = { c0, c1, c2, c3 };
#pragma unroll
    for (int t = 0; t < 4; ++t) {
        const int col = n0 + t * 16 + n;
        const float bval = bias[col];
#pragma unroll
        for (int i = 0; i < 8; ++i)
            out[(size_t)(m0 + i + 8 * half) * Dm + col] = cc[t][i] + bval;
    }
}

// ---------------------------------------------------------------------------
extern "C" void kernel_launch(void* const* d_in, const int* in_sizes, int n_in,
                              void* d_out, int out_size, void* d_ws, size_t ws_size,
                              hipStream_t stream) {
    (void)in_sizes; (void)n_in; (void)out_size; (void)ws_size;

    const float* x  = (const float*)d_in[0];
    const float* y  = (const float*)d_in[1];
    const float* wq = (const float*)d_in[2];
    const float* bq = (const float*)d_in[3];
    const float* wk = (const float*)d_in[4];
    const float* bk = (const float*)d_in[5];
    const float* wv = (const float*)d_in[6];
    const float* bv = (const float*)d_in[7];
    const float* wo = (const float*)d_in[8];
    const float* bo = (const float*)d_in[9];
    float* out = (float*)d_out;

    const size_t nQKV = (size_t)Mtot * Dm;   // 8M bf16 elements per tensor
    __bf16* qws = (__bf16*)d_ws;
    __bf16* kws = qws + nQKV;
    __bf16* vws = kws + nQKV;
    __bf16* aws = vws + nQKV;

    const dim3 blk(256);
    // 64 m-blocks x 16 n-strips = 1024 blocks; z selects Q/K/V
    proj_qkv_kernel<<<dim3(1024, 1, 3), blk, 0, stream>>>(
        x, y, wq, bq, wk, bk, wv, bv, qws, kws, vws);
    // 64 (b,h) x 16 query-strips = 1024 blocks
    attn_kernel<<<dim3(1024), blk, 0, stream>>>(qws, kws, vws, aws);
    oproj_kernel<<<dim3(1024), blk, 0, stream>>>(aws, wo, bo, out);
}
